// LSTMLM100M_34806414967470
// MI455X (gfx1250) — compile-verified, matrix operand order checked by reading, not run on hardware
//
#include <hip/hip_runtime.h>
#include <stdint.h>
#include <stddef.h>

// ---------------- problem constants ----------------
#define DMODEL 640
#define HDIM   1760
#define VOCAB  50257
#define BB     8
#define SS     512
#define G4H    (4 * HDIM)      // 7040
#define MROWS  (BB * SS)       // 4096

typedef __attribute__((ext_vector_type(16))) __bf16 v16bf;
typedef __attribute__((ext_vector_type(8)))  float  v8f;

#define AS1 __attribute__((address_space(1)))
#define AS3 __attribute__((address_space(3)))

#if defined(__has_builtin)
#  if __has_builtin(__builtin_amdgcn_global_load_async_to_lds_b128)
#    define USE_ASYNC_BUILTIN 1
#  endif
#  if __has_builtin(__builtin_amdgcn_s_wait_asynccnt)
#    define USE_WAIT_BUILTIN 1
#  endif
#endif

// exact pointee type the builtin expects: int __attribute__((vector_size(16)))
typedef int v4i_vs __attribute__((vector_size(16)));

// 16-byte global -> LDS async copy (ASYNCcnt-tracked, no VGPR staging)
__device__ __forceinline__ void async_cp16(__bf16* ldst, const __bf16* gsrc) {
#if defined(USE_ASYNC_BUILTIN)
    void* g = (void*)gsrc;   // drop const, keep generic
    void* l = (void*)ldst;
    __builtin_amdgcn_global_load_async_to_lds_b128(
        (AS1 v4i_vs*)g,
        (AS3 v4i_vs*)l, 0, 0);
#else
    unsigned loff = (unsigned)(uintptr_t)(AS3 void*)(void*)ldst;
    asm volatile("global_load_async_to_lds_b128 %0, %1, off"
                 :: "v"(loff), "v"(gsrc) : "memory");
#endif
}

#if defined(USE_WAIT_BUILTIN)
#  define WAIT_ASYNC(n) __builtin_amdgcn_s_wait_asynccnt(n)
#else
#  define WAIT_ASYNC(n) asm volatile("s_wait_asynccnt %0" :: "i"(n) : "memory")
#endif

union FragU { uint4 u[2]; v16bf v; };

// Per-lane bf16 WMMA fragment load (A 16x32 or mirrored B 32x16 layout).
// lane 0-15: row = lane,     K chunks [0..7]  and [16..23]
// lane 16-31: row = lane-16, K chunks [8..15] and [24..31]
__device__ __forceinline__ v16bf load_frag(const __bf16* tile, int ld, int lane) {
    const int row = lane & 15;
    const int kb  = (lane & 16) ? 8 : 0;
    FragU f;
    f.u[0] = *(const uint4*)(tile + (size_t)row * ld + kb);
    f.u[1] = *(const uint4*)(tile + (size_t)row * ld + kb + 16);
    return f.v;
}

__device__ __forceinline__ float sigm(float x) {
    return 1.0f / (1.0f + __expf(-x));
}

// ---------------- elementwise kernels ----------------
__global__ void cvt_f32_bf16(const float* __restrict__ in, __bf16* __restrict__ out, size_t n) {
    size_t i = (size_t)blockIdx.x * blockDim.x + threadIdx.x;
    size_t stride = (size_t)gridDim.x * blockDim.x;
    for (; i < n; i += stride) out[i] = (__bf16)in[i];
}

__global__ void embed_gather(const int* __restrict__ ids, const float* __restrict__ emb,
                             __bf16* __restrict__ out) {
    size_t i = (size_t)blockIdx.x * blockDim.x + threadIdx.x;
    if (i >= (size_t)MROWS * DMODEL) return;
    int row = (int)(i / DMODEL);
    int d   = (int)(i % DMODEL);
    out[i] = (__bf16)emb[(size_t)ids[row] * DMODEL + d];
}

__global__ void zero_state(float* __restrict__ c, __bf16* __restrict__ h) {
    int i = blockIdx.x * blockDim.x + threadIdx.x;
    if (i < BB * HDIM)  c[i] = 0.0f;
    if (i < 16 * HDIM)  h[i] = (__bf16)0.0f;   // 16 rows: 8 batch + 8 zero pad
}

// ---------------- big tiled GEMM: C[M,N] = A[M,K] * B[N,K]^T (+bias) ----------------
// A,B bf16 row-major (K contiguous). M multiple of 128, K multiple of 32. N guarded.
// Double-buffered async global->LDS staging (ASYNCcnt pipelined).
template <bool STORE_BF16>
__global__ __launch_bounds__(256)
void gemm_bf16_wmma(const __bf16* __restrict__ A, const __bf16* __restrict__ B,
                    void* __restrict__ Cv,
                    const float* __restrict__ bias0, const float* __restrict__ bias1,
                    int M, int N, int K) {
    __shared__ __align__(16) __bf16 As[2][128 * 32];
    __shared__ __align__(16) __bf16 Bs[2][128 * 32];

    const int tid  = threadIdx.x;
    const int lane = tid & 31;
    const int wave = __builtin_amdgcn_readfirstlane(tid >> 5);  // uniform wave id 0..7
    const int wm   = (wave & 1) * 64;   // wave M origin in tile
    const int wn   = (wave >> 1) * 32;  // wave N origin in tile
    const int m0   = blockIdx.y * 128;
    const int n0   = blockIdx.x * 128;

    v8f acc[4][2];
    for (int mi = 0; mi < 4; ++mi)
        for (int ni = 0; ni < 2; ++ni)
            for (int e = 0; e < 8; ++e) acc[mi][ni][e] = 0.0f;

    // stage one 128x32 slab of A and B into LDS buffer `buf` (4 async b128/thread)
    auto stage = [&](int buf, int k0) {
#pragma unroll
        for (int i = 0; i < 2; ++i) {
            int idx = tid + i * 256;        // 0..511
            int r   = idx >> 2;             // 0..127
            int c   = (idx & 3) * 8;        // 0,8,16,24
            async_cp16(&As[buf][r * 32 + c], &A[(size_t)(m0 + r) * K + k0 + c]);
            int gr = n0 + r; if (gr >= N) gr = N - 1;
            async_cp16(&Bs[buf][r * 32 + c], &B[(size_t)gr * K + k0 + c]);
        }
    };

    const int KT = K >> 5;
    stage(0, 0);

    for (int kt = 0; kt < KT; ++kt) {
        const int cur = kt & 1;
        if (kt + 1 < KT) {
            stage(cur ^ 1, (kt + 1) * 32);   // prefetch next slab into other buffer
            // deep cache hint two slabs ahead
            if (kt + 2 < KT) {
                int pr = tid & 127;
                __builtin_prefetch(&A[(size_t)(m0 + pr) * K + (kt + 2) * 32], 0, 1);
            }
            WAIT_ASYNC(4);                   // retire current slab (in-order), next stays in flight
        } else {
            WAIT_ASYNC(0);
        }
        __syncthreads();

        v16bf af[4], bfr[2];
#pragma unroll
        for (int mi = 0; mi < 4; ++mi) af[mi]  = load_frag(&As[cur][(wm + mi * 16) * 32], 32, lane);
#pragma unroll
        for (int ni = 0; ni < 2; ++ni) bfr[ni] = load_frag(&Bs[cur][(wn + ni * 16) * 32], 32, lane);

#pragma unroll
        for (int mi = 0; mi < 4; ++mi)
#pragma unroll
            for (int ni = 0; ni < 2; ++ni)
                acc[mi][ni] = __builtin_amdgcn_wmma_f32_16x16x32_bf16(
                    false, af[mi], false, bfr[ni], (short)0, acc[mi][ni], false, false);
        __syncthreads();   // all waves done with `cur` before it is overwritten next iter
    }

    // epilogue: acc vgpr r -> row (lane>=16 ? r+8 : r), col lane&15
    const int rb    = (lane >> 4) * 8;
    const int col_l = lane & 15;
#pragma unroll
    for (int mi = 0; mi < 4; ++mi)
#pragma unroll
        for (int ni = 0; ni < 2; ++ni)
#pragma unroll
            for (int r = 0; r < 8; ++r) {
                int row = m0 + wm + mi * 16 + rb + r;
                int col = n0 + wn + ni * 16 + col_l;
                if (col < N) {
                    float v = acc[mi][ni][r];
                    if (bias0) v += bias0[col];
                    if (bias1) v += bias1[col];
                    if (STORE_BF16) ((__bf16*)Cv)[(size_t)row * N + col] = (__bf16)v;
                    else            ((float*)Cv)[(size_t)row * N + col]  = v;
                }
            }
}

// ---------------- recurrent step GEMM: gbuf[8,4H] = xg[:,t,:] + h[8,H] @ w_hh[4H,H]^T ----
// h stored as 16 rows (rows 8..15 zero) so M pads exactly to one WMMA tile.
// 8 waves / block: 4 N-tiles x 2-way split-K (uniform 27-iter loops + tail),
// partials reduced through LDS.
__global__ __launch_bounds__(256)
void lstm_step_gemm(const __bf16* __restrict__ hbf,   // [16, H]
                    const __bf16* __restrict__ whh,   // [4H, H] bf16
                    const float*  __restrict__ xg,    // [B, S, 4H] fp32
                    float* __restrict__ gbuf,         // [8, 4H]
                    int t) {
    __shared__ float sred[4 * 8 * 32];                // [tile][vgpr][lane]
    const int lane  = threadIdx.x & 31;
    const int wave  = __builtin_amdgcn_readfirstlane(threadIdx.x >> 5);  // uniform 0..7
    const int tile  = wave >> 1;                      // 0..3
    const int ks    = wave & 1;                       // K half
    const int nbase = (blockIdx.x * 4 + tile) * 16;   // 440 tiles total

    v8f acc;
    for (int e = 0; e < 8; ++e) acc[e] = 0.0f;

    const __bf16* brow = whh + (size_t)nbase * HDIM;
    // 55 K-iters: both halves run a constant 27-trip (unrollable) loop;
    // the ks==0 wave takes iteration 54 as a uniform-branch tail.
    const int base = ks * 27;
#pragma unroll 9
    for (int i = 0; i < 27; ++i) {
        const int k0 = (base + i) * 32;
        v16bf a = load_frag(hbf + k0, HDIM, lane);
        v16bf b = load_frag(brow + k0, HDIM, lane);
        acc = __builtin_amdgcn_wmma_f32_16x16x32_bf16(
            false, a, false, b, (short)0, acc, false, false);
    }
    if (ks == 0) {
        const int k0 = 54 * 32;
        v16bf a = load_frag(hbf + k0, HDIM, lane);
        v16bf b = load_frag(brow + k0, HDIM, lane);
        acc = __builtin_amdgcn_wmma_f32_16x16x32_bf16(
            false, a, false, b, (short)0, acc, false, false);
    }

    if (ks == 1) {
#pragma unroll
        for (int r = 0; r < 8; ++r) sred[((tile * 8) + r) * 32 + lane] = acc[r];
    }
    __syncthreads();
    if (ks == 0) {
#pragma unroll
        for (int r = 0; r < 8; ++r) acc[r] += sred[((tile * 8) + r) * 32 + lane];

        const int rb  = (lane >> 4) * 8;
        const int col = nbase + (lane & 15);
#pragma unroll
        for (int r = 0; r < 8; ++r) {
            int row = rb + r;                 // 0..15; only real batch rows stored
            if (row < BB)
                gbuf[(size_t)row * G4H + col] =
                    acc[r] + xg[((size_t)row * SS + t) * G4H + col];
        }
    }
}

// ---------------- LSTM gate elementwise ----------------
__global__ void lstm_gate(const float* __restrict__ gbuf, float* __restrict__ c,
                          __bf16* __restrict__ h_state,   // [16, H] (pad rows stay 0)
                          __bf16* __restrict__ h_seq,     // [B, S, H]
                          int t) {
    int i = blockIdx.x * blockDim.x + threadIdx.x;
    if (i >= BB * HDIM) return;
    int b = i / HDIM, j = i % HDIM;
    const float* g = gbuf + (size_t)b * G4H;
    float gi = g[j], gf = g[HDIM + j], gg = g[2 * HDIM + j], go = g[3 * HDIM + j];
    float cn = sigm(gf) * c[i] + sigm(gi) * tanhf(gg);
    float hn = sigm(go) * tanhf(cn);
    c[i] = cn;
    h_state[(size_t)b * HDIM + j] = (__bf16)hn;
    h_seq[((size_t)b * SS + t) * HDIM + j] = (__bf16)hn;
}

// ---------------- LayerNorm over H ----------------
__global__ __launch_bounds__(256)
void layernorm_bf16(const __bf16* __restrict__ x, const float* __restrict__ g,
                    const float* __restrict__ b, __bf16* __restrict__ y) {
    __shared__ float s1[256], s2[256];
    const size_t base = (size_t)blockIdx.x * HDIM;
    const int tid = threadIdx.x;
    float s = 0.0f, q = 0.0f;
    for (int j = tid; j < HDIM; j += 256) {
        float v = (float)x[base + j];
        s += v; q += v * v;
    }
    s1[tid] = s; s2[tid] = q;
    __syncthreads();
    for (int st = 128; st > 0; st >>= 1) {
        if (tid < st) { s1[tid] += s1[tid + st]; s2[tid] += s2[tid + st]; }
        __syncthreads();
    }
    float mu  = s1[0] * (1.0f / HDIM);
    float var = s2[0] * (1.0f / HDIM) - mu * mu;
    float rs  = rsqrtf(var + 1e-5f);
    for (int j = tid; j < HDIM; j += 256) {
        float v = (float)x[base + j];
        y[base + j] = (__bf16)((v - mu) * rs * g[j] + b[j]);
    }
}

// ---------------- host orchestration ----------------
extern "C" void kernel_launch(void* const* d_in, const int* in_sizes, int n_in,
                              void* d_out, int out_size, void* d_ws, size_t ws_size,
                              hipStream_t stream) {
    (void)in_sizes; (void)n_in; (void)out_size; (void)ws_size;

    const int*   ids    = (const int*)d_in[0];
    const float* emb    = (const float*)d_in[1];
    const float* w_ih[3] = {(const float*)d_in[2],  (const float*)d_in[6],  (const float*)d_in[10]};
    const float* w_hh[3] = {(const float*)d_in[3],  (const float*)d_in[7],  (const float*)d_in[11]};
    const float* b_ih[3] = {(const float*)d_in[4],  (const float*)d_in[8],  (const float*)d_in[12]};
    const float* b_hh[3] = {(const float*)d_in[5],  (const float*)d_in[9],  (const float*)d_in[13]};
    const float* ln_g   = (const float*)d_in[14];
    const float* ln_b   = (const float*)d_in[15];
    const float* proj_w = (const float*)d_in[16];

    // -------- workspace carve (~350 MB) --------
    char* p = (char*)d_ws;
    auto carve = [&](size_t bytes) -> void* {
        void* r = (void*)p;
        p += (bytes + 255) & ~(size_t)255;
        return r;
    };
    const size_t n_wih[3] = {(size_t)G4H * DMODEL, (size_t)G4H * HDIM, (size_t)G4H * HDIM};
    __bf16* wih_bf[3]; __bf16* whh_bf[3];
    for (int l = 0; l < 3; ++l) wih_bf[l] = (__bf16*)carve(n_wih[l] * 2);
    for (int l = 0; l < 3; ++l) whh_bf[l] = (__bf16*)carve((size_t)G4H * HDIM * 2);
    __bf16* emb_bf  = (__bf16*)carve((size_t)VOCAB * DMODEL * 2);
    __bf16* proj_bf = (__bf16*)carve((size_t)DMODEL * HDIM * 2);
    __bf16* bufA    = (__bf16*)carve((size_t)MROWS * HDIM * 2);
    __bf16* bufB    = (__bf16*)carve((size_t)MROWS * HDIM * 2);
    float*  xg      = (float*) carve((size_t)MROWS * G4H * 4);
    float*  gbuf    = (float*) carve((size_t)BB * G4H * 4);
    __bf16* hstate  = (__bf16*)carve((size_t)16 * HDIM * 2);
    float*  cstate  = (float*) carve((size_t)BB * HDIM * 4);
    __bf16* pout    = (__bf16*)carve((size_t)MROWS * DMODEL * 2);

    // -------- weight conversions fp32 -> bf16 --------
    auto conv = [&](const float* in, __bf16* out, size_t n) {
        cvt_f32_bf16<<<2048, 256, 0, stream>>>(in, out, n);
    };
    for (int l = 0; l < 3; ++l) conv(w_ih[l], wih_bf[l], n_wih[l]);
    for (int l = 0; l < 3; ++l) conv(w_hh[l], whh_bf[l], (size_t)G4H * HDIM);
    conv(emb,    emb_bf,  (size_t)VOCAB * DMODEL);
    conv(proj_w, proj_bf, (size_t)DMODEL * HDIM);

    // -------- embedding gather --------
    {
        size_t n = (size_t)MROWS * DMODEL;
        embed_gather<<<(unsigned)((n + 255) / 256), 256, 0, stream>>>(ids, emb, bufA);
    }

    // -------- 3 LSTM layers --------
    __bf16* in_act  = bufA;
    __bf16* out_act = bufB;
    const int Kl[3] = {DMODEL, HDIM, HDIM};
    for (int l = 0; l < 3; ++l) {
        // gate pre-GEMM: xg = act @ w_ih^T + (b_ih + b_hh)
        dim3 gg(G4H / 128, MROWS / 128);
        gemm_bf16_wmma<false><<<gg, 256, 0, stream>>>(
            in_act, wih_bf[l], xg, b_ih[l], b_hh[l], MROWS, G4H, Kl[l]);

        zero_state<<<(16 * HDIM + 255) / 256, 256, 0, stream>>>(cstate, hstate);

        for (int t = 0; t < SS; ++t) {
            lstm_step_gemm<<<G4H / 64, 256, 0, stream>>>(hstate, whh_bf[l], xg, gbuf, t);
            lstm_gate<<<(BB * HDIM + 255) / 256, 256, 0, stream>>>(gbuf, cstate, hstate, out_act, t);
        }
        __bf16* tmp = in_act; in_act = out_act; out_act = tmp;
    }
    // final LSTM output now in in_act

    // -------- LayerNorm --------
    layernorm_bf16<<<MROWS, 256, 0, stream>>>(in_act, ln_g, ln_b, out_act);

    // -------- projection to d_model (bf16 out) --------
    {
        dim3 gp(DMODEL / 128, MROWS / 128);
        gemm_bf16_wmma<true><<<gp, 256, 0, stream>>>(
            out_act, proj_bf, pout, nullptr, nullptr, MROWS, DMODEL, HDIM);
    }

    // -------- tied-embedding logits into d_out (fp32) --------
    {
        dim3 gl((VOCAB + 127) / 128, MROWS / 128);
        gemm_bf16_wmma<false><<<gl, 256, 0, stream>>>(
            pout, emb_bf, d_out, nullptr, nullptr, MROWS, VOCAB, DMODEL);
    }
}
